// Column1_20298015441326
// MI455X (gfx1250) — matrix operand order repaired and checked by reading, not compiled, not run on hardware
//
#include <hip/hip_runtime.h>

// ---------------------------------------------------------------------------
// MI455X (gfx1250) branch-local GEMM + kWTA.
//   pot[r] = rec[:,0,r,:] (32x2048) @ W[r,:,0,:]^T (2048x32)
// 512 independent 32x32x2048 GEMMs -> memory bound (AI ~= 8 FLOP/B,
// 270 MB @ 23.3 TB/s ~= 11.5 us floor). Compute made free by splitting
// fp32 into bf16 hi/lo and using 3x v_wmma_f32_16x16x32_bf16 per K-step
// (error ~2^-16 relative; bf16 WMMA rate >> fp32 K=4 WMMA rate).
// All global traffic is touched exactly once -> non-temporal loads/stores
// (TH=NT) so the one-pass stream does not churn the 192 MB L2.
// One block (4 wave32) per branch; each wave owns one 16x16 tile.
// ---------------------------------------------------------------------------

#define K_FEAT   32
#define T_STEPS  32
#define RF_N     512
#define L_LEN    2048
#define THRESHV  20.0f
#define KWTA     4

#define CH   64              // L-chunk per pipeline stage
#define NCH  (L_LEN / CH)    // 32 stages
#define SA   (CH + 8)        // padded LDS row stride (bf16 elems) -> bank spread

typedef __bf16 bf16_t;
typedef __attribute__((ext_vector_type(16))) bf16_t v16bf;
typedef __attribute__((ext_vector_type(8)))  bf16_t v8bf;   // 16 B
typedef __attribute__((ext_vector_type(4)))  bf16_t v4bf;   // 8 B
typedef __attribute__((ext_vector_type(8)))  float  v8f;
typedef __attribute__((ext_vector_type(4)))  float  v4f;    // 16 B, NT-loadable

union FragU { v16bf v; v8bf h[2]; };

__global__ __launch_bounds__(128) void branch_gemm_kwta(
    const float* __restrict__ rec,   // (T, 1, RF, L)
    const float* __restrict__ W,     // (RF, K, 1, L)
    float* __restrict__ out)         // (T, 1, K, RF)
{
  __shared__ bf16_t Ahi[32 * SA];
  __shared__ bf16_t Alo[32 * SA];
  __shared__ bf16_t Bhi[32 * SA];
  __shared__ bf16_t Blo[32 * SA];
  __shared__ float  pot[32 * 33];    // 32x32 potentials, padded
  __shared__ float  maskS[K_FEAT];

  const int r    = blockIdx.x;
  const int tid  = threadIdx.x;
  const int lane = tid & 31;
  const int wave = tid >> 5;
  const int half = lane >> 4;        // hi/lo 16-lane group
  const int i16  = lane & 15;
  const int mi   = wave >> 1;        // tile row (0..1) -> M 0..15 / 16..31
  const int ni   = wave & 1;         // tile col (0..1) -> N 0..15 / 16..31

  const float* Abase = rec + (size_t)r * L_LEN;            // row t: + t*RF*L
  const float* Bbase = W   + (size_t)r * K_FEAT * L_LEN;   // row k contiguous

  // ---- software pipeline: NT 128-bit global loads one chunk ahead ---------
  v4f ra[4], rb[4];
#pragma unroll
  for (int j = 0; j < 4; ++j) {
    int idx = j * 128 + tid;         // 512 float4 per 32x64 tile
    int row = idx >> 4, col = idx & 15;
    ra[j] = __builtin_nontemporal_load(
        (const v4f*)(Abase + (size_t)row * (RF_N * L_LEN) + col * 4));
    rb[j] = __builtin_nontemporal_load(
        (const v4f*)(Bbase + (size_t)row * L_LEN + col * 4));
  }

  v8f acc = {};

  for (int c = 0; c < NCH; ++c) {
    __syncthreads();                                   // prev compute done
    // ---- convert f32 -> bf16 hi/lo and stage into LDS ---------------------
#pragma unroll
    for (int j = 0; j < 4; ++j) {
      int idx = j * 128 + tid;
      int row = idx >> 4, col = idx & 15;
      v4bf ah, al, bh, bl;
#pragma unroll
      for (int e = 0; e < 4; ++e) {
        float xa = ra[j][e], xb = rb[j][e];
        bf16_t h = (bf16_t)xa;
        ah[e] = h;  al[e] = (bf16_t)(xa - (float)h);
        h = (bf16_t)xb;
        bh[e] = h;  bl[e] = (bf16_t)(xb - (float)h);
      }
      int o = row * SA + col * 4;                      // 8 B aligned
      *(v4bf*)&Ahi[o] = ah;  *(v4bf*)&Alo[o] = al;
      *(v4bf*)&Bhi[o] = bh;  *(v4bf*)&Blo[o] = bl;
    }
    __syncthreads();                                   // tiles visible

    if (c + 1 < NCH) {                                 // issue next loads early
      int l0 = (c + 1) * CH;
#pragma unroll
      for (int j = 0; j < 4; ++j) {
        int idx = j * 128 + tid;
        int row = idx >> 4, col = idx & 15;
        ra[j] = __builtin_nontemporal_load(
            (const v4f*)(Abase + (size_t)row * (RF_N * L_LEN) + l0 + col * 4));
        rb[j] = __builtin_nontemporal_load(
            (const v4f*)(Bbase + (size_t)row * L_LEN + l0 + col * 4));
      }
    }

    // ---- 16x16x32 bf16 WMMA, split-precision (3 per K-step) ---------------
    const int aoff = (mi * 16 + i16) * SA;   // A row m (16-bit A layout 16x32)
    const int boff = (ni * 16 + i16) * SA;   // B col n (16-bit B layout 32x16)
#pragma unroll
    for (int k0 = 0; k0 < CH; k0 += 32) {
      FragU ah, al, bh, bl;
      // A frag: V0-3 = K[8h..8h+7], V4-7 = K[16+8h..16+8h+7]
      ah.h[0] = *(const v8bf*)&Ahi[aoff + k0 + 8 * half];
      ah.h[1] = *(const v8bf*)&Ahi[aoff + k0 + 16 + 8 * half];
      al.h[0] = *(const v8bf*)&Alo[aoff + k0 + 8 * half];
      al.h[1] = *(const v8bf*)&Alo[aoff + k0 + 16 + 8 * half];
      // B frag: V0-3 = K[8h..8h+7], V4-7 = K[16+8h..16+8h+7]
      bh.h[0] = *(const v8bf*)&Bhi[boff + k0 + 8 * half];
      bh.h[1] = *(const v8bf*)&Bhi[boff + k0 + 16 + 8 * half];
      bl.h[0] = *(const v8bf*)&Blo[boff + k0 + 8 * half];
      bl.h[1] = *(const v8bf*)&Blo[boff + k0 + 16 + 8 * half];

      acc = __builtin_amdgcn_wmma_f32_16x16x32_bf16(
                false, al.v, false, bh.v, (short)0, acc, false, false);
      acc = __builtin_amdgcn_wmma_f32_16x16x32_bf16(
                false, ah.v, false, bl.v, (short)0, acc, false, false);
      acc = __builtin_amdgcn_wmma_f32_16x16x32_bf16(
                false, ah.v, false, bh.v, (short)0, acc, false, false);
    }
  }

  // ---- spill C tiles: VGPR v -> (M = v + 8*half, N = i16) per ISA layout --
#pragma unroll
  for (int v = 0; v < 8; ++v)
    pot[(mi * 16 + half * 8 + v) * 33 + ni * 16 + i16] = acc[v];
  __syncthreads();

  // ---- kWTA (wave 0, lane = feature k) ------------------------------------
  if (wave == 0) {
    const int k = lane;
    float nsp = 0.f;
#pragma unroll
    for (int t = 0; t < T_STEPS; ++t)
      nsp += (pot[t * 33 + k] > THRESHV) ? 1.f : 0.f;
    int first = T_STEPS - (int)nsp;                 // clip(32 - n, 0, 31)
    if (first > 31) first = 31;
    float p  = pot[first * 33 + k];
    float vf = (p > THRESHV) ? p : 0.f;             // thr at first spike
    float cand = (nsp > 0.f) ? vf : 0.f;            // max_t trunc[t,k]
    float mx = cand;
#pragma unroll
    for (int off = 16; off; off >>= 1)
      mx = fmaxf(mx, __shfl_xor(mx, off, 32));
    float tot = nsp * (vf + 32.f * mx);             // sum_t(trunc + spikes*v)

    float m = 0.f;
#pragma unroll
    for (int it = 0; it < KWTA; ++it) {             // top-4, low-index ties
      float bv = tot; int bi = k;
#pragma unroll
      for (int off = 16; off; off >>= 1) {
        float ov = __shfl_xor(bv, off, 32);
        int   oi = __shfl_xor(bi, off, 32);
        if (ov > bv || (ov == bv && oi < bi)) { bv = ov; bi = oi; }
      }
      if (k == bi) { if (tot > 0.f) m = 1.f; tot = -1.f; }
    }
    maskS[k] = m;
  }
  __syncthreads();

  // ---- out[t,0,k,r] = sign(thr * mask) = spike & mask ---------------------
#pragma unroll
  for (int j = 0; j < 8; ++j) {
    int e = j * 128 + tid;
    int t = e >> 5, k = e & 31;
    float s = (pot[t * 33 + k] > THRESHV && maskS[k] > 0.f) ? 1.f : 0.f;
    __builtin_nontemporal_store(s, &out[(size_t)(t * K_FEAT + k) * RF_N + r]);
  }
}

extern "C" void kernel_launch(void* const* d_in, const int* in_sizes, int n_in,
                              void* d_out, int out_size, void* d_ws, size_t ws_size,
                              hipStream_t stream) {
  (void)in_sizes; (void)n_in; (void)d_ws; (void)ws_size; (void)out_size;
  const float* rec = (const float*)d_in[0];   // (32,1,512,2048) f32
  const float* W   = (const float*)d_in[1];   // (512,32,1,2048) f32
  // d_in[2] (reward) is unused by the reference computation.
  float* out = (float*)d_out;                 // (32,1,32,512)   f32
  branch_gemm_kwta<<<RF_N, 128, 0, stream>>>(rec, W, out);
}
